// GraphLearner_88373247082872
// MI455X (gfx1250) — matrix operand order repaired and checked by reading, not compiled, not run on hardware
//
#include <hip/hip_runtime.h>
#include <hip/hip_fp16.h>

typedef _Float16 v16h __attribute__((ext_vector_type(16)));
typedef float    v8f  __attribute__((ext_vector_type(8)));

#define NB      4      // batch
#define NN      2048   // nodes
#define HID     64
#define NHEADS  4
#define DH      16
#define TSTEPS  32
#define DDYN    16
#define DSTAT   8
#define TOPK    10
#define ALPHA   0.3f

// ---------------------------------------------------------------------------
// Kernel 1: fused encoders + q/k projection, emits f16 q,k with 1/sqrt(dh)
// folded into q. One 64-thread block per (b,n) row.
// ---------------------------------------------------------------------------
__global__ __launch_bounds__(64) void enc_qk_kernel(
    const float* __restrict__ x_dyn, const float* __restrict__ x_stat,
    const float* __restrict__ W_dyn, const float* __restrict__ b_dyn,
    const float* __restrict__ W_stat, const float* __restrict__ b_stat,
    const float* __restrict__ W_fuse, const float* __restrict__ b_fuse,
    const float* __restrict__ Wq, const float* __restrict__ Wk,
    _Float16* __restrict__ qout, _Float16* __restrict__ kout)
{
    __shared__ float xbar[DDYN];
    __shared__ float xs[DSTAT];
    __shared__ float hbuf[2 * HID];
    __shared__ float hf[HID];

    const int row = blockIdx.x;          // b*NN + n
    const int j   = threadIdx.x;         // 0..63 output channel

    if (j < DDYN) {
        const float* p = x_dyn + ((size_t)row * TSTEPS) * DDYN + j;
        float s = 0.f;
        #pragma unroll
        for (int t = 0; t < TSTEPS; ++t) s += p[t * DDYN];
        xbar[j] = s * (1.f / TSTEPS);
    }
    if (j < DSTAT) xs[j] = x_stat[(size_t)row * DSTAT + j];
    __syncthreads();

    float hd = b_dyn[j];
    #pragma unroll
    for (int d = 0; d < DDYN; ++d) hd = fmaf(xbar[d], W_dyn[d * HID + j], hd);
    hd = fmaxf(hd, 0.f);

    float hs = b_stat[j];
    #pragma unroll
    for (int d = 0; d < DSTAT; ++d) hs = fmaf(xs[d], W_stat[d * HID + j], hs);
    hs = fmaxf(hs, 0.f);

    hbuf[j] = hd;
    hbuf[HID + j] = hs;
    __syncthreads();

    float f = b_fuse[j];
    #pragma unroll 8
    for (int d = 0; d < 2 * HID; ++d) f = fmaf(hbuf[d], W_fuse[d * HID + j], f);
    hf[j] = f;
    __syncthreads();

    float qv = 0.f, kv = 0.f;
    #pragma unroll 8
    for (int d = 0; d < HID; ++d) {
        float hv = hf[d];
        qv = fmaf(hv, Wq[d * HID + j], qv);
        kv = fmaf(hv, Wk[d * HID + j], kv);
    }
    qv *= 0.25f;  // fold 1/sqrt(dh), dh=16
    qout[(size_t)row * HID + j] = (_Float16)qv;
    kout[(size_t)row * HID + j] = (_Float16)kv;
}

// ---------------------------------------------------------------------------
// Kernel 2: multi-head softmax(QK^T) averaged over heads -> A_learn.
// Block = (b, 16-row stripe), 4 waves. The whole K matrix for batch b
// (2048x64 f16 = 256KB) is staged in LDS via async global->LDS loads;
// wave w stages and owns head w for the two stat passes (row max, exp-sum),
// so softmax stats are computed entirely in-wave. Pass 3 switches to
// col-chunk ownership (512 cols/wave, all 4 heads from LDS) for a single
// accumulated global write of the head-averaged probabilities.
//
// WMMA f32_16x16x32_f16 with K padded 16->32: A lanes 16..31 are zero, so
// B lanes 16..31 (K=16..31) may carry ANY finite data -> they duplicate
// lanes 0..15 (same LDS address, broadcast), eliminating all divergence.
// ---------------------------------------------------------------------------
__global__ __launch_bounds__(128) void attn_kernel(
    const _Float16* __restrict__ qmat, const _Float16* __restrict__ kmat,
    float* __restrict__ Alearn)
{
    // per-head K panel: ksh[h][col*DH + d] = K[b, col, h*16 + d]
    __shared__ __align__(16) _Float16 ksh[NHEADS][NN * DH];   // 256 KB
    __shared__ float smh[NHEADS][16];                         // row max
    __shared__ float slh[NHEADS][16];                         // 0.25/row-sum

    const int rb   = blockIdx.x;        // 0..127 row stripe
    const int b    = blockIdx.y;
    const int tid  = threadIdx.x;
    const int w    = tid >> 5;          // wave id == head id for passes 1-2
    const int lane = tid & 31;
    const int half = lane >> 4;
    const int lr   = lane & 15;

    union HV { v16h v; uint4 u[2]; };

    // ---- preload A fragments (Q rows) for all 4 heads ------------------
    // A 16x32 f16: lane&15 = M row; lanes<16 carry K=0..15 in halves 0..15,
    // lanes>=16 (K=16..31) are zero padding.
    HV afr[NHEADS];
    {
        const int row = rb * 16 + lr;
        const _Float16* qp = qmat + ((size_t)(b * NN + row)) * HID;
        const int dbase = half * 8;
        #pragma unroll
        for (int h = 0; h < NHEADS; ++h) {
            afr[h].u[0] = half ? make_uint4(0u, 0u, 0u, 0u) : *(const uint4*)(qp + h * DH);
            afr[h].u[1] = half ? make_uint4(0u, 0u, 0u, 0u) : *(const uint4*)(qp + h * DH + 8);
        }
    }

    // ---- stage head w's K panel (all 2048 cols) via async loads --------
    const _Float16* kbase = kmat + ((size_t)b * NN) * HID;
    {
        const unsigned lbase = (unsigned)(size_t)(&ksh[w][0]);
        // 2048 cols * 32B = 64KB per wave; 16B per lane per issue
        for (int it = 0; it < 128; ++it) {
            const int e   = it * 32 + lane;   // 0..4095
            const int col = e >> 1;
            const int seg = e & 1;            // which 16B half of the col
            const unsigned long long ga =
                (unsigned long long)(kbase + (size_t)col * HID + w * DH + seg * 8);
            const unsigned la = lbase + (unsigned)(col * 32 + seg * 16);
            asm volatile("global_load_async_to_lds_b128 %0, %1, off"
                         :: "v"(la), "v"(ga) : "memory");
        }
        asm volatile("s_wait_asynccnt 0x0" ::: "memory");
    }
    __syncthreads();

    // ---- pass 1: row max for head w over all 2048 cols -----------------
    float mloc[8];
    #pragma unroll
    for (int i = 0; i < 8; ++i) mloc[i] = -1e30f;
    for (int t = 0; t < 128; ++t) {
        const uint4* kp = (const uint4*)&ksh[w][(t * 16 + lr) * DH];
        HV bfr;
        bfr.u[0] = kp[0];
        bfr.u[1] = kp[1];
        v8f c = {0.f, 0.f, 0.f, 0.f, 0.f, 0.f, 0.f, 0.f};
        c = __builtin_amdgcn_wmma_f32_16x16x32_f16(false, afr[w].v,
                                                   false, bfr.v,
                                                   (short)0, c, false, false);
        #pragma unroll
        for (int i = 0; i < 8; ++i) mloc[i] = fmaxf(mloc[i], c[i]);
    }
    #pragma unroll
    for (int mask = 1; mask < 16; mask <<= 1)
        #pragma unroll
        for (int i = 0; i < 8; ++i)
            mloc[i] = fmaxf(mloc[i], __shfl_xor(mloc[i], mask, 32));

    // ---- pass 2: sum of exp(s - m) for head w --------------------------
    float ls[8];
    #pragma unroll
    for (int i = 0; i < 8; ++i) ls[i] = 0.f;
    for (int t = 0; t < 128; ++t) {
        const uint4* kp = (const uint4*)&ksh[w][(t * 16 + lr) * DH];
        HV bfr;
        bfr.u[0] = kp[0];
        bfr.u[1] = kp[1];
        v8f c = {0.f, 0.f, 0.f, 0.f, 0.f, 0.f, 0.f, 0.f};
        c = __builtin_amdgcn_wmma_f32_16x16x32_f16(false, afr[w].v,
                                                   false, bfr.v,
                                                   (short)0, c, false, false);
        #pragma unroll
        for (int i = 0; i < 8; ++i) ls[i] += __expf(c[i] - mloc[i]);
    }
    #pragma unroll
    for (int mask = 1; mask < 16; mask <<= 1)
        #pragma unroll
        for (int i = 0; i < 8; ++i) ls[i] += __shfl_xor(ls[i], mask, 32);

    if (lr == 0) {
        #pragma unroll
        for (int i = 0; i < 8; ++i) {
            smh[w][half * 8 + i] = mloc[i];
            slh[w][half * 8 + i] = 0.25f / ls[i];   // fold head-mean here
        }
    }
    __syncthreads();

    // ---- load all heads' stats for this lane's 8 rows ------------------
    float mreg[NHEADS][8], lreg[NHEADS][8];
    #pragma unroll
    for (int h = 0; h < NHEADS; ++h)
        #pragma unroll
        for (int i = 0; i < 8; ++i) {
            mreg[h][i] = smh[h][half * 8 + i];
            lreg[h][i] = slh[h][half * 8 + i];
        }

    // ---- pass 3: head-averaged probabilities, col chunk per wave -------
    const int c0 = w * 512;
    for (int t = 0; t < 32; ++t) {
        const int col = c0 + t * 16 + lr;
        float acc[8];
        #pragma unroll
        for (int i = 0; i < 8; ++i) acc[i] = 0.f;
        #pragma unroll
        for (int h = 0; h < NHEADS; ++h) {
            const uint4* kp = (const uint4*)&ksh[h][col * DH];
            HV bfr;
            bfr.u[0] = kp[0];
            bfr.u[1] = kp[1];
            v8f c = {0.f, 0.f, 0.f, 0.f, 0.f, 0.f, 0.f, 0.f};
            c = __builtin_amdgcn_wmma_f32_16x16x32_f16(false, afr[h].v,
                                                       false, bfr.v,
                                                       (short)0, c, false, false);
            #pragma unroll
            for (int i = 0; i < 8; ++i)
                acc[i] += __expf(c[i] - mreg[h][i]) * lreg[h][i];
        }
        #pragma unroll
        for (int i = 0; i < 8; ++i) {
            const int row = rb * 16 + half * 8 + i;
            Alearn[((size_t)(b * NN + row)) * NN + col] = acc[i];
        }
    }
}

// ---------------------------------------------------------------------------
// Kernel 3: top-10 per row of A_learn -> compact (idx, val) lists.
// 128 threads/row; row cached in registers; 10x argmax with LDS tree reduce.
// ---------------------------------------------------------------------------
__global__ __launch_bounds__(128) void topk_kernel(
    const float* __restrict__ Alearn,
    int* __restrict__ tidx, float* __restrict__ tval)
{
    __shared__ float sv[128];
    __shared__ int   si[128];

    const int row = blockIdx.x;     // b*NN + n
    const int t   = threadIdx.x;
    const float* ap = Alearn + (size_t)row * NN;

    float v[16];
    #pragma unroll
    for (int u = 0; u < 16; ++u) v[u] = ap[t + u * 128];

    for (int ksel = 0; ksel < TOPK; ++ksel) {
        float best = -1e30f; int bu = 0;
        #pragma unroll
        for (int u = 0; u < 16; ++u)
            if (v[u] > best) { best = v[u]; bu = u; }
        sv[t] = best;
        si[t] = t + bu * 128;
        __syncthreads();
        for (int s = 64; s > 0; s >>= 1) {
            if (t < s) {
                if (sv[t + s] > sv[t] ||
                    (sv[t + s] == sv[t] && si[t + s] < si[t])) {
                    sv[t] = sv[t + s]; si[t] = si[t + s];
                }
            }
            __syncthreads();
        }
        const int   wcol = si[0];
        const float wval = sv[0];
        if (t == 0) {
            tidx[(size_t)row * TOPK + ksel] = wcol;
            tval[(size_t)row * TOPK + ksel] = wval;
        }
        if ((wcol & 127) == t) v[wcol >> 7] = -1e30f;   // remove winner
        __syncthreads();
    }
}

// ---------------------------------------------------------------------------
// Kernel 4: A_final = a*prior + 0.35*(As[i,j] + As[j,i]) via compact lists;
// row-sum -> degree; writes A_final AND L in one pass (row staged in LDS).
// ---------------------------------------------------------------------------
__global__ __launch_bounds__(256) void final_kernel(
    const float* __restrict__ prior,
    const int* __restrict__ tidx, const float* __restrict__ tval,
    float* __restrict__ L, float* __restrict__ Afin)
{
    __shared__ float afrow[NN];        // 8 KB
    __shared__ float rsum[256];
    __shared__ int   li[TOPK];
    __shared__ float lv[TOPK];

    const int b = blockIdx.y;
    const int i = blockIdx.x;
    const size_t rowb = (size_t)(b * NN + i);
    const int t = threadIdx.x;

    if (t < TOPK) {
        li[t] = tidx[rowb * TOPK + t];
        lv[t] = tval[rowb * TOPK + t];
    }
    __syncthreads();

    float sum = 0.f;
    #pragma unroll
    for (int u = 0; u < 8; ++u) {
        const int j = t + u * 256;
        float gi = 0.f;
        #pragma unroll
        for (int s = 0; s < TOPK; ++s) gi = (li[s] == j) ? lv[s] : gi;
        float gj = 0.f;
        const size_t rj = (size_t)(b * NN + j) * TOPK;
        #pragma unroll
        for (int s = 0; s < TOPK; ++s) {
            const int   id = tidx[rj + s];
            const float vv = tval[rj + s];
            gj = (id == i) ? vv : gj;
        }
        const float af = ALPHA * prior[rowb * NN + j]
                       + 0.5f * (1.f - ALPHA) * (gi + gj);
        afrow[j] = af;
        sum += af;
    }
    rsum[t] = sum;
    __syncthreads();
    for (int s = 128; s > 0; s >>= 1) {
        if (t < s) rsum[t] += rsum[t + s];
        __syncthreads();
    }
    const float degree = rsum[0];

    #pragma unroll
    for (int u = 0; u < 8; ++u) {
        const int j = t + u * 256;
        const float af = afrow[j];
        Afin[rowb * NN + j] = af;
        L[rowb * NN + j]    = ((j == i) ? degree : 0.f) - af;
    }
}

// ---------------------------------------------------------------------------
extern "C" void kernel_launch(void* const* d_in, const int* in_sizes, int n_in,
                              void* d_out, int out_size, void* d_ws, size_t ws_size,
                              hipStream_t stream)
{
    const float* x_dyn  = (const float*)d_in[0];
    const float* x_stat = (const float*)d_in[1];
    const float* A_prior= (const float*)d_in[2];
    const float* W_dyn  = (const float*)d_in[3];
    const float* b_dyn  = (const float*)d_in[4];
    const float* W_stat = (const float*)d_in[5];
    const float* b_stat = (const float*)d_in[6];
    const float* W_fuse = (const float*)d_in[7];
    const float* b_fuse = (const float*)d_in[8];
    const float* Wq     = (const float*)d_in[9];
    const float* Wk     = (const float*)d_in[10];

    // Workspace layout (~2.7 MB total):
    //   q:    NB*NN*HID f16 = 1 MB
    //   k:    NB*NN*HID f16 = 1 MB
    //   tidx: NB*NN*TOPK i32 = 320 KB
    //   tval: NB*NN*TOPK f32 = 320 KB
    char* ws = (char*)d_ws;
    _Float16* q = (_Float16*)ws;
    _Float16* k = (_Float16*)(ws + (1u << 20));
    int*   tidx = (int*)  (ws + (2u << 20));
    float* tval = (float*)(ws + (2u << 20) + (size_t)NB * NN * TOPK * 4);

    float* L    = (float*)d_out;
    float* Afin = L + (size_t)NB * NN * NN;
    float* Astage = L;   // stage A_learn in the (later-overwritten) L slot

    enc_qk_kernel<<<NB * NN, 64, 0, stream>>>(
        x_dyn, x_stat, W_dyn, b_dyn, W_stat, b_stat,
        W_fuse, b_fuse, Wq, Wk, q, k);

    attn_kernel<<<dim3(NN / 16, NB), 128, 0, stream>>>(q, k, Astage);

    topk_kernel<<<NB * NN, 128, 0, stream>>>(Astage, tidx, tval);

    final_kernel<<<dim3(NN, NB), 256, 0, stream>>>(A_prior, tidx, tval, L, Afin);
}
